// ZeroSAttention_57389353009768
// MI455X (gfx1250) — compile-verified, matrix operand order checked by reading, not run on hardware
//
#include <hip/hip_runtime.h>
#include <hip/hip_bf16.h>

#define L_SEQ 1024
#define D_MODEL 768
#define NH 12
#define HD 64

typedef __attribute__((ext_vector_type(16))) __bf16 v16bf;
typedef __attribute__((ext_vector_type(8)))  __bf16 v8bf;
typedef __attribute__((ext_vector_type(8)))  float  v8f;

__device__ __forceinline__ v8f wmma_bf16(v16bf a, v16bf b, v8f c) {
    // D = A(16x32) * B(32x16) + C, f32 accumulate
    return __builtin_amdgcn_wmma_f32_16x16x32_bf16(
        false, a, false, b, (short)0, c, false, false);
}

// Load an A/B fragment for one 32-wide K chunk from a row-major row.
// Lane layout: element e<8 -> k = koff+e ; e>=8 -> k = 16+koff+(e-8),
// where koff = 8*(lane>=16) is already folded into p by the caller.
__device__ __forceinline__ v16bf load_frag(const __bf16* p) {
    v8bf lo = *(const v8bf*)(p);
    v8bf hi = *(const v8bf*)(p + 16);
    v16bf r;
#pragma unroll
    for (int i = 0; i < 8; ++i) { r[i] = lo[i]; r[i + 8] = hi[i]; }
    return r;
}

// ---------------------------------------------------------------- converts
__global__ void f2bf_kernel(const float* __restrict__ s, __bf16* __restrict__ d, int n) {
    int i = blockIdx.x * blockDim.x + threadIdx.x;
    if (i < n) d[i] = (__bf16)s[i];
}

// v (L,H,d) f32 -> per-head transposed bf16: Vt[h][n][i] = v[i][h*64+n]
__global__ void vtrans_kernel(const float* __restrict__ v, __bf16* __restrict__ vt) {
    int idx = blockIdx.x * blockDim.x + threadIdx.x;   // over NH*HD*L
    if (idx >= NH * HD * L_SEQ) return;
    int i = idx & (L_SEQ - 1);
    int hn = idx >> 10;          // h*64+n
    vt[idx] = (__bf16)v[(size_t)i * D_MODEL + hn];
}

// ---------------------------------------------------------------- WMMA GEMM
// Y[M,N] = Xb[M,K] @ Wb[N,K]^T + bias[N]   (f32 out)
// one wave computes a 16x64 output strip; 4 waves / block
__global__ void gemm_xwT_bf16(const __bf16* __restrict__ X, const __bf16* __restrict__ W,
                              const float* __restrict__ bias, float* __restrict__ Y,
                              int M, int N, int K) {
    int wave = blockIdx.x * (blockDim.x >> 5) + (threadIdx.x >> 5);
    int lane = threadIdx.x & 31;
    int ntiles = N >> 6;
    int mt = wave / ntiles, nt = wave % ntiles;
    if (mt >= (M >> 4)) return;
    int m0 = mt << 4, n0 = nt << 6;
    int lrow = lane & 15;
    int koff = (lane >> 4) << 3;   // 0 or 8

    v8f acc[4] = {};
    const __bf16* xrow = X + (size_t)(m0 + lrow) * K + koff;
    for (int k0 = 0; k0 < K; k0 += 32) {
        v16bf a = load_frag(xrow + k0);
#pragma unroll
        for (int j = 0; j < 4; ++j) {
            const __bf16* wrow = W + (size_t)(n0 + (j << 4) + lrow) * K + koff;
            v16bf b = load_frag(wrow + k0);
            acc[j] = wmma_bf16(a, b, acc[j]);
        }
    }
    int mbase = m0 + ((lane >> 4) << 3);
#pragma unroll
    for (int j = 0; j < 4; ++j) {
        int n = n0 + (j << 4) + lrow;
        float bv = bias ? bias[n] : 0.0f;
#pragma unroll
        for (int r = 0; r < 8; ++r)
            Y[(size_t)(mbase + r) * N + n] = acc[j][r] + bv;
    }
}

// ---------------------------------------------------------------- gate GEMM (tiny: 1024x36)
__global__ void gate_kernel(const float* __restrict__ x, const float* __restrict__ Wg,
                            const float* __restrict__ bg, float* __restrict__ gate) {
    int idx = blockIdx.x * blockDim.x + threadIdx.x;
    if (idx >= L_SEQ * 36) return;
    int l = idx / 36, g = idx % 36;
    const float* xr = x + (size_t)l * D_MODEL;
    const float* wr = Wg + (size_t)g * D_MODEL;
    float acc = bg[g];
    for (int k = 0; k < D_MODEL; ++k) acc += xr[k] * wr[k];
    gate[idx] = acc;
}

// ---------------------------------------------------------------- s_i scan
// one wave per head; lanes cover 2 dims each; sequential over L
__global__ void si_kernel(const float* __restrict__ U, const float* __restrict__ mu,
                          const float* __restrict__ plt, float* __restrict__ sArr) {
    int h = blockIdx.x;
    int lane = threadIdx.x;   // 32 threads
    float tau = expf(fminf(fmaxf(plt[h], -50.0f), 30.0f));
    float c0 = tau * mu[h * HD + lane];
    float c1 = tau * mu[h * HD + lane + 32];
    float a0 = 0.0f, a1 = 0.0f;
    for (int l = 0; l < L_SEQ; ++l) {
        const float* u = U + (size_t)l * D_MODEL + h * HD;
        float u0 = u[lane], u1 = u[lane + 32];
        a0 += u0; a1 += u1;
        float t = (float)(l + 1);
        float inv = 1.0f / (tau + t);
        float p = u0 * ((c0 + a0) * inv) + u1 * ((c1 + a1) * inv);
#pragma unroll
        for (int o = 16; o > 0; o >>= 1) p += __shfl_xor(p, o, 32);
        if (lane == 0) sArr[l * NH + h] = -p * 0.125f;   // /sqrt(64)
    }
}

// ---------------------------------------------------------------- RoPE + L2 norm -> bf16
// one wave per (l,h); lanes handle dims j=lane and j=lane+32
__global__ void rope_l2_kernel(const float* __restrict__ Q, __bf16* __restrict__ Qb) {
    int wv = blockIdx.x * (blockDim.x >> 5) + (threadIdx.x >> 5);
    if (wv >= L_SEQ * NH) return;
    int l = wv / NH, h = wv % NH;
    int lane = threadIdx.x & 31;
    const float* q = Q + (size_t)l * D_MODEL + h * HD;
    float x0 = q[lane], x1 = q[lane + 32];
    // cos/sin index: freq f(j) = j/2 ; rotate_half over 32/32 split
    float f0 = (float)(lane >> 1);
    float ang0 = (float)l * __powf(10000.0f, -2.0f * f0 / 64.0f);
    float ang1 = (float)l * __powf(10000.0f, -2.0f * (f0 + 16.0f) / 64.0f);
    float r0 = x0 * __cosf(ang0) - x1 * __sinf(ang0);
    float r1 = x1 * __cosf(ang1) + x0 * __sinf(ang1);
    float ss = r0 * r0 + r1 * r1;
#pragma unroll
    for (int o = 16; o > 0; o >>= 1) ss += __shfl_xor(ss, o, 32);
    float n = fmaxf(sqrtf(ss), 1e-12f);
    __bf16* qb = Qb + (size_t)l * D_MODEL + h * HD;
    qb[lane]      = (__bf16)(r0 / n);
    qb[lane + 32] = (__bf16)(r1 / n);
}

// ---------------------------------------------------------------- E_t/P_t scan -> alpha/beta/gamma
__global__ void scan_kernel(const float* __restrict__ sArr, const float* __restrict__ gate,
                            float* __restrict__ eArr, float* __restrict__ al,
                            float* __restrict__ be, float* __restrict__ ga) {
    int h = blockIdx.x;
    __shared__ float red[256];
    float m = -3.4e38f;
    for (int l = threadIdx.x; l < L_SEQ; l += 256) m = fmaxf(m, sArr[l * NH + h]);
    red[threadIdx.x] = m;
    __syncthreads();
    for (int s = 128; s > 0; s >>= 1) {
        if (threadIdx.x < s) red[threadIdx.x] = fmaxf(red[threadIdx.x], red[threadIdx.x + s]);
        __syncthreads();
    }
    float smax = red[0];
    for (int l = threadIdx.x; l < L_SEQ; l += 256)
        eArr[l * NH + h] = expf(sArr[l * NH + h] - smax);
    __syncthreads();
    if (threadIdx.x == 0) {
        float E = 0.0f, P = 0.0f;
        for (int l = 0; l < L_SEQ; ++l) {
            float si = sArr[l * NH + h];
            E += eArr[l * NH + h];
            P += si;
            float g1 = gate[l * 36 + h * 3 + 0];
            float gh = gate[l * 36 + h * 3 + 1];
            float s1 = 1.0f / (1.0f + expf(-g1));
            float sh = 1.0f / (1.0f + expf(-gh));
            float t = (float)(l + 1);
            float a = sh / (E + 1e-12f);
            float b = (s1 - sh) / t;
            float g = -((b * P + sh) / t);
            al[l * NH + h] = a;
            be[l * NH + h] = b;
            ga[l * NH + h] = g;
        }
    }
}

// ---------------------------------------------------------------- WMMA causal attention
// out[t] = sum_{i<=t} (q_t.k_i) * (alpha_t*exp_s_i + beta_t*s_i + gamma_t) * v_i
// one wave per (head, 16-row t-tile); 4 waves / block
// vt is per-head transposed V: vt[h][n][i], so S.V B-frags are contiguous loads.
__global__ void attn_kernel(const __bf16* __restrict__ qb, const __bf16* __restrict__ kb,
                            const __bf16* __restrict__ vt,
                            const float* __restrict__ sArr, const float* __restrict__ eArr,
                            const float* __restrict__ al, const float* __restrict__ be,
                            const float* __restrict__ ga, float* __restrict__ O) {
    __shared__ __align__(16) __bf16 sm[4][16][40];   // per-wave scratch, padded rows
    int wv = threadIdx.x >> 5;
    int lane = threadIdx.x & 31;
    int gw = blockIdx.x * 4 + wv;
    int tTiles = L_SEQ >> 4;
    int h = gw / tTiles;
    int tt = gw % tTiles;
    if (h >= NH) return;
    int m0 = tt << 4;
    int lrow = lane & 15;
    int koff = (lane >> 4) << 3;
    int mbase = m0 + ((lane >> 4) << 3);

    // loop-invariant per-row coefficients
    float alr[8], ber[8], gar[8];
#pragma unroll
    for (int r = 0; r < 8; ++r) {
        int t = mbase + r;
        alr[r] = al[t * NH + h];
        ber[r] = be[t * NH + h];
        gar[r] = ga[t * NH + h];
    }

    const __bf16* Qrow = qb + (size_t)(m0 + lrow) * D_MODEL + h * HD + koff;
    v16bf aq0 = load_frag(Qrow);        // k-dims 0..31
    v16bf aq1 = load_frag(Qrow + 32);   // k-dims 32..63

    v8f o[4] = {};
    for (int i0 = 0; i0 <= m0 + 15; i0 += 32) {
        // S tile (16 t-rows x 32 i-cols) = Q Kt^T over d=64
        v8f s[2] = {{}, {}};
#pragma unroll
        for (int nn = 0; nn < 2; ++nn) {
            const __bf16* Krow = kb + (size_t)(i0 + nn * 16 + lrow) * D_MODEL + h * HD + koff;
            v16bf b0 = load_frag(Krow);
            v16bf b1 = load_frag(Krow + 32);
            s[nn] = wmma_bf16(aq0, b0, s[nn]);
            s[nn] = wmma_bf16(aq1, b1, s[nn]);
        }
        // weight + causal mask, D-frag -> LDS (as 16x32 row-major bf16)
#pragma unroll
        for (int nn = 0; nn < 2; ++nn) {
#pragma unroll
            for (int r = 0; r < 8; ++r) {
                int t = mbase + r;
                int i = i0 + nn * 16 + lrow;
                float w = 0.0f;
                if (i <= t)
                    w = alr[r] * eArr[i * NH + h] + ber[r] * sArr[i * NH + h] + gar[r];
                sm[wv][mbase - m0 + r][nn * 16 + lrow] = (__bf16)(s[nn][r] * w);
            }
        }
        // reload as A-frag (LDS ops in-order within wave)
        v16bf as = load_frag(&sm[wv][lrow][koff]);
        // O(16x64) += A(16x32) @ V(32x64); B[k][n] = vt[h][n][i0+k]
#pragma unroll
        for (int j = 0; j < 4; ++j) {
            int n = j * 16 + lrow;
            const __bf16* vrow = vt + ((size_t)(h * HD + n) << 10) + i0 + koff;
            v16bf bvf = load_frag(vrow);
            o[j] = wmma_bf16(as, bvf, o[j]);
        }
    }
#pragma unroll
    for (int j = 0; j < 4; ++j)
#pragma unroll
        for (int r = 0; r < 8; ++r)
            O[(size_t)(mbase + r) * D_MODEL + h * HD + j * 16 + lrow] = o[j][r];
}

// ---------------------------------------------------------------- per-head LayerNorm -> bf16
__global__ void ln_kernel(const float* __restrict__ O, __bf16* __restrict__ Ob) {
    int wv = blockIdx.x * (blockDim.x >> 5) + (threadIdx.x >> 5);
    if (wv >= L_SEQ * NH) return;
    int l = wv / NH, h = wv % NH;
    int lane = threadIdx.x & 31;
    const float* o = O + (size_t)l * D_MODEL + h * HD;
    float a = o[lane], b = o[lane + 32];
    float s = a + b;
#pragma unroll
    for (int t = 16; t > 0; t >>= 1) s += __shfl_xor(s, t, 32);
    float mean = s * (1.0f / 64.0f);
    float da = a - mean, db = b - mean;
    float v = da * da + db * db;
#pragma unroll
    for (int t = 16; t > 0; t >>= 1) v += __shfl_xor(v, t, 32);
    float rstd = rsqrtf(v * (1.0f / 64.0f) + 1e-5f);
    __bf16* ob = Ob + (size_t)l * D_MODEL + h * HD;
    ob[lane]      = (__bf16)(da * rstd);
    ob[lane + 32] = (__bf16)(db * rstd);
}

// ---------------------------------------------------------------- launcher
extern "C" void kernel_launch(void* const* d_in, const int* in_sizes, int n_in,
                              void* d_out, int out_size, void* d_ws, size_t ws_size,
                              hipStream_t stream) {
    const float* x   = (const float*)d_in[0];
    const float* Wq  = (const float*)d_in[1];
    const float* bq  = (const float*)d_in[2];
    const float* Wk  = (const float*)d_in[3];
    const float* bk  = (const float*)d_in[4];
    const float* Wv  = (const float*)d_in[5];
    const float* bv  = (const float*)d_in[6];
    const float* Wu  = (const float*)d_in[7];
    const float* bu  = (const float*)d_in[8];
    const float* Wg  = (const float*)d_in[9];
    const float* bg  = (const float*)d_in[10];
    const float* Wo  = (const float*)d_in[11];
    const float* bo  = (const float*)d_in[12];
    const float* pmu = (const float*)d_in[13];
    const float* plt = (const float*)d_in[14];

    const size_t LD = (size_t)L_SEQ * D_MODEL;    // 786432
    const size_t DD = (size_t)D_MODEL * D_MODEL;  // 589824
    const size_t LH = (size_t)L_SEQ * NH;         // 12288

    char* w = (char*)d_ws;
    size_t off = 0;
    auto alloc = [&](size_t bytes) -> char* {
        off = (off + 255) & ~(size_t)255;
        char* p = w + off;
        off += bytes;
        return p;
    };
    __bf16* xb  = (__bf16*)alloc(LD * 2);
    __bf16* wqb = (__bf16*)alloc(DD * 2);
    __bf16* wkb = (__bf16*)alloc(DD * 2);
    __bf16* wvb = (__bf16*)alloc(DD * 2);
    __bf16* wub = (__bf16*)alloc(DD * 2);
    __bf16* wob = (__bf16*)alloc(DD * 2);
    float*  qf  = (float*)alloc(LD * 4);
    float*  kf  = (float*)alloc(LD * 4);
    float*  vf  = (float*)alloc(LD * 4);
    float*  uf  = (float*)alloc(LD * 4);
    __bf16* qbb = (__bf16*)alloc(LD * 2);
    __bf16* kbb = (__bf16*)alloc(LD * 2);
    __bf16* vtb = (__bf16*)alloc(LD * 2);          // per-head transposed V
    float*  gatef = (float*)alloc((size_t)L_SEQ * 36 * 4);
    float*  sArr  = (float*)alloc(LH * 4);
    float*  eArr  = (float*)alloc(LH * 4);
    float*  al    = (float*)alloc(LH * 4);
    float*  be    = (float*)alloc(LH * 4);
    float*  ga    = (float*)alloc(LH * 4);
    float*  Of    = (float*)alloc(LD * 4);
    __bf16* obb   = (__bf16*)alloc(LD * 2);
    (void)ws_size; (void)n_in; (void)in_sizes; (void)out_size;

    // 1. bf16 casts
    f2bf_kernel<<<(int)((LD + 255) / 256), 256, 0, stream>>>(x,  xb,  (int)LD);
    f2bf_kernel<<<(int)((DD + 255) / 256), 256, 0, stream>>>(Wq, wqb, (int)DD);
    f2bf_kernel<<<(int)((DD + 255) / 256), 256, 0, stream>>>(Wk, wkb, (int)DD);
    f2bf_kernel<<<(int)((DD + 255) / 256), 256, 0, stream>>>(Wv, wvb, (int)DD);
    f2bf_kernel<<<(int)((DD + 255) / 256), 256, 0, stream>>>(Wu, wub, (int)DD);
    f2bf_kernel<<<(int)((DD + 255) / 256), 256, 0, stream>>>(Wo, wob, (int)DD);

    // 2. projections: 64 m-tiles * 12 n-strips = 768 waves -> 192 blocks of 128
    gemm_xwT_bf16<<<192, 128, 0, stream>>>(xb, wqb, bq, qf, L_SEQ, D_MODEL, D_MODEL);
    gemm_xwT_bf16<<<192, 128, 0, stream>>>(xb, wkb, bk, kf, L_SEQ, D_MODEL, D_MODEL);
    gemm_xwT_bf16<<<192, 128, 0, stream>>>(xb, wvb, bv, vf, L_SEQ, D_MODEL, D_MODEL);
    gemm_xwT_bf16<<<192, 128, 0, stream>>>(xb, wub, bu, uf, L_SEQ, D_MODEL, D_MODEL);

    // 3. gate projection (1024 x 36)
    gate_kernel<<<(L_SEQ * 36 + 255) / 256, 256, 0, stream>>>(x, Wg, bg, gatef);

    // 4. v -> per-head transposed bf16
    vtrans_kernel<<<(int)((LD + 255) / 256), 256, 0, stream>>>(vf, vtb);

    // 5. s_i radial logits
    si_kernel<<<NH, 32, 0, stream>>>(uf, pmu, plt, sArr);

    // 6. RoPE + L2 norm for q, k
    rope_l2_kernel<<<(int)(LH / 4), 128, 0, stream>>>(qf, qbb);
    rope_l2_kernel<<<(int)(LH / 4), 128, 0, stream>>>(kf, kbb);

    // 7. E_t/P_t scans -> alpha/beta/gamma
    scan_kernel<<<NH, 256, 0, stream>>>(sArr, gatef, eArr, al, be, ga);

    // 8. causal WMMA attention: 12 heads * 64 t-tiles = 768 waves -> 192 blocks
    attn_kernel<<<192, 128, 0, stream>>>(qbb, kbb, vtb, sArr, eArr, al, be, ga, Of);

    // 9. per-head LayerNorm -> bf16
    ln_kernel<<<(int)(LH / 4), 128, 0, stream>>>(Of, obb);

    // 10. output projection -> d_out (f32)
    gemm_xwT_bf16<<<192, 128, 0, stream>>>(obb, wob, bo, (float*)d_out,
                                           L_SEQ, D_MODEL, D_MODEL);
}
